// BEVTransformer_66417374266009
// MI455X (gfx1250) — matrix operand order repaired and checked by reading, not compiled, not run on hardware
//
#include <hip/hip_runtime.h>
#include <hip/hip_bf16.h>
#include <math.h>

#define DEVFN __device__ __forceinline__

typedef __attribute__((ext_vector_type(16))) _Float16 v16h;
typedef __attribute__((ext_vector_type(8)))  float    v8f;

// ---- problem constants (match reference) ----
constexpr int BEV = 64;
constexpr int NQ = BEV * BEV;           // 4096
constexpr int EMBED = 256;
constexpr int NUM_CAMS = 6;
constexpr int NUM_HEADS = 8;
constexpr int NUM_VALUE = 7729;
constexpr int OUT_DIMS = 80;
constexpr int LVL_H[4] = {58, 29, 15, 8};
constexpr int LVL_W[4] = {100, 50, 25, 13};
constexpr int LVL_S[4] = {0, 5800, 7250, 7625};

// =====================================================================
// Generic f16-WMMA GEMM:  out = act(A[M,K] @ W[N,K]^T + bias + addend)
// One wave computes a 32x32 tile (2x2 WMMA fragments), f32 accumulate.
// =====================================================================
DEVFN v16h load_a_frag(const float* __restrict__ A, int lda, int M, int mbase, int kbase) {
  const int lane = threadIdx.x & 31;
  const int half = lane >> 4;                 // K-half select per ISA layout
  int m = mbase + (lane & 15);
  m = (m < M) ? m : (M - 1);                  // clamped load; store is guarded
  const float* r = A + (size_t)m * lda + kbase + half * 8;
  v16h f;
#pragma unroll
  for (int e = 0; e < 8; ++e) f[e] = (_Float16)r[e];          // K = half*8 + e
#pragma unroll
  for (int e = 0; e < 8; ++e) f[8 + e] = (_Float16)r[16 + e]; // K = 16 + half*8 + e
  return f;
}

DEVFN v16h load_b_frag(const float* __restrict__ W, int ldw, int N, int nbase, int kbase) {
  const int lane = threadIdx.x & 31;
  int n = nbase + (lane & 15);
  n = (n < N) ? n : (N - 1);
  const float* r = W + (size_t)n * ldw + kbase + (lane >> 4) * 8;
  v16h f;
#pragma unroll
  for (int e = 0; e < 8; ++e) f[e] = (_Float16)r[e];          // K = half*8 + e
#pragma unroll
  for (int e = 0; e < 8; ++e) f[8 + e] = (_Float16)r[16 + e]; // K = 16 + half*8 + e
  return f;
}

__global__ void k_gemm(const float* __restrict__ A, const float* __restrict__ W,
                       const float* __restrict__ bias, const float* __restrict__ addend,
                       float* __restrict__ out,
                       int M, int N, int K, int relu, int tstore) {
  const int wid = blockIdx.x * (blockDim.x >> 5) + (threadIdx.x >> 5);
  const int nt = (N + 31) >> 5;
  const int mt = (M + 31) >> 5;
  if (wid >= mt * nt) return;                  // whole-wave exit (EXEC stays all-1 for WMMA)
  const int m0 = (wid / nt) * 32;
  const int n0 = (wid % nt) * 32;

  v8f acc00 = {}, acc01 = {}, acc10 = {}, acc11 = {};
  for (int k = 0; k < K; k += 32) {            // K is a multiple of 32 for all our GEMMs
    v16h a0 = load_a_frag(A, K, M, m0,      k);
    v16h a1 = load_a_frag(A, K, M, m0 + 16, k);
    v16h b0 = load_b_frag(W, K, N, n0,      k);
    v16h b1 = load_b_frag(W, K, N, n0 + 16, k);
    acc00 = __builtin_amdgcn_wmma_f32_16x16x32_f16(false, a0, false, b0, (short)0, acc00, false, false);
    acc01 = __builtin_amdgcn_wmma_f32_16x16x32_f16(false, a0, false, b1, (short)0, acc01, false, false);
    acc10 = __builtin_amdgcn_wmma_f32_16x16x32_f16(false, a1, false, b0, (short)0, acc10, false, false);
    acc11 = __builtin_amdgcn_wmma_f32_16x16x32_f16(false, a1, false, b1, (short)0, acc11, false, false);
  }

  const int lane = threadIdx.x & 31;
  const int mloc = (lane >> 4) << 3;           // C/D layout: M = r + 8*(lane>=16)
  const int nloc = lane & 15;
#pragma unroll
  for (int i = 0; i < 2; ++i) {
#pragma unroll
    for (int j = 0; j < 2; ++j) {
      const v8f acc = (i == 0) ? ((j == 0) ? acc00 : acc01) : ((j == 0) ? acc10 : acc11);
      const int n = n0 + j * 16 + nloc;
#pragma unroll
      for (int r = 0; r < 8; ++r) {
        const int m = m0 + i * 16 + mloc + r;
        if (m < M && n < N) {
          float v = acc[r];
          if (bias)   v += bias[n];
          if (addend) v += addend[(size_t)m * N + n];
          if (relu)   v = fmaxf(v, 0.f);
          if (tstore) out[(size_t)n * M + m] = v;   // column-major store (final transpose)
          else        out[(size_t)m * N + n] = v;
        }
      }
    }
  }
}

// =====================================================================
// Positional MLP layer 1: pe128 = relu(pos2d @ w1^T + b1), K=2 (VALU)
// =====================================================================
__global__ void k_posmlp1(const float* __restrict__ w1, const float* __restrict__ b1,
                          float* __restrict__ pe128) {
  const int idx = blockIdx.x * blockDim.x + threadIdx.x;
  if (idx >= NQ * 128) return;
  const int q = idx >> 7, j = idx & 127;
  const float gx = (float)(q & 63) * (1.f / 63.f);
  const float gy = (float)(q >> 6) * (1.f / 63.f);
  const float v = w1[j * 2] * gx + w1[j * 2 + 1] * gy + b1[j];
  pe128[idx] = fmaxf(v, 0.f);
}

// =====================================================================
// Reference-point projection per (cam, query): 4x4 matvec per z-slice.
// Writes refc[cam][q][z][2] (u,v) and maskq[cam][q] in {0,1}.
// =====================================================================
__global__ void k_refcam(const float* __restrict__ l2i,
                         float* __restrict__ refc, float* __restrict__ maskq) {
  const int idx = blockIdx.x * blockDim.x + threadIdx.x;
  if (idx >= NUM_CAMS * NQ) return;
  const int cam = idx / NQ, q = idx % NQ;
  const float* Mx = l2i + cam * 16;
  const float xs = ((float)(q & 63) + 0.5f) * (1.f / 64.f);
  const float ys = ((float)(q >> 6) + 0.5f) * (1.f / 64.f);
  const float X = xs * 102.4f - 51.2f;
  const float Y = ys * 102.4f - 51.2f;
  float any = 0.f;
#pragma unroll
  for (int z = 0; z < 4; ++z) {
    const float zs = (0.5f + (float)z * (7.f / 3.f)) * (1.f / 8.f); // linspace(.5,7.5,4)/8
    const float Zc = zs * 8.f - 5.f;
    const float c0 = Mx[0] * X + Mx[1] * Y + Mx[2]  * Zc + Mx[3];
    const float c1 = Mx[4] * X + Mx[5] * Y + Mx[6]  * Zc + Mx[7];
    const float c2 = Mx[8] * X + Mx[9] * Y + Mx[10] * Zc + Mx[11];
    const float d  = fmaxf(c2, 1e-5f);
    const float u  = (c0 / d) * (1.f / 1600.f);
    const float v  = (c1 / d) * (1.f / 928.f);
    const bool ok = (c2 > 1e-5f) && (u > 0.f) && (u < 1.f) && (v > 0.f) && (v < 1.f);
    refc[((size_t)idx * 4 + z) * 2 + 0] = u;
    refc[((size_t)idx * 4 + z) * 2 + 1] = v;
    any += ok ? 1.f : 0.f;
  }
  maskq[idx] = (any > 0.f) ? 1.f : 0.f;
}

// =====================================================================
// Softmax over 32 (level,point) logits per (query, head). One wave each.
// =====================================================================
__global__ void k_softmax(float* __restrict__ aw) {
  const int wid = blockIdx.x * (blockDim.x >> 5) + (threadIdx.x >> 5);
  if (wid >= NQ * NUM_HEADS) return;
  const int lane = threadIdx.x & 31;
  float* p = aw + (size_t)wid * 32;
  const float v = p[lane];
  float mx = v;
#pragma unroll
  for (int m = 16; m; m >>= 1) mx = fmaxf(mx, __shfl_xor(mx, m, 32));
  const float e = __expf(v - mx);
  float s = e;
#pragma unroll
  for (int m = 16; m; m >>= 1) s += __shfl_xor(s, m, 32);
  p[lane] = e / s;
}

// =====================================================================
// Deformable sampling core. One wave per (cam, head, query); lane = channel.
// Each bilinear tap is one coalesced 128B L2-resident load across the wave.
// Skips (cam,q) pairs with empty mask (their attn is never read).
// =====================================================================
__global__ void k_sample(const float* __restrict__ v, const float* __restrict__ refc,
                         const float* __restrict__ maskq, const float* __restrict__ off,
                         const float* __restrict__ aw, float* __restrict__ attn) {
  const int wid = blockIdx.x * (blockDim.x >> 5) + (threadIdx.x >> 5);
  if (wid >= NUM_CAMS * NQ * NUM_HEADS) return;
  const int h   = wid & 7;
  const int q   = (wid >> 3) & (NQ - 1);
  const int cam = wid >> 15;                    // NQ*NUM_HEADS == 1<<15
  if (maskq[cam * NQ + q] == 0.f) return;       // wave-uniform skip

  const int lane = threadIdx.x & 31;
  const float* __restrict__ vb = v + (size_t)cam * NUM_VALUE * EMBED + h * 32 + lane;
  const float* rp = refc + (size_t)(cam * NQ + q) * 8;
  float rx[4], ry[4];
#pragma unroll
  for (int z = 0; z < 4; ++z) { rx[z] = rp[z * 2]; ry[z] = rp[z * 2 + 1]; }
  const float* op = off + (size_t)q * 512 + h * 64;   // (h,l,p,2)
  const float* ap = aw  + (size_t)q * 256 + h * 32;   // (h,l,p)

  float acc = 0.f;
#pragma unroll
  for (int l = 0; l < 4; ++l) {
    const int Hl = LVL_H[l], Wl = LVL_W[l], s0 = LVL_S[l];
    const float fW = (float)Wl, fH = (float)Hl;
    const float iW = 1.f / fW, iH = 1.f / fH;
#pragma unroll
    for (int p = 0; p < 8; ++p) {
      const int z = p & 3;                      // points flattened as (group=2, z=4)
      const float x = (rx[z] + op[(l * 8 + p) * 2 + 0] * iW) * fW - 0.5f;
      const float y = (ry[z] + op[(l * 8 + p) * 2 + 1] * iH) * fH - 0.5f;
      const float x0f = floorf(x), y0f = floorf(y);
      const float wx = x - x0f, wy = y - y0f;
      const int x0 = (int)x0f, y0 = (int)y0f;
      const int x1 = x0 + 1, y1 = y0 + 1;
      const bool okx0 = (x0 >= 0) && (x0 < Wl), oky0 = (y0 >= 0) && (y0 < Hl);
      const bool okx1 = (x1 >= 0) && (x1 < Wl), oky1 = (y1 >= 0) && (y1 < Hl);
      const int cx0 = min(max(x0, 0), Wl - 1), cy0 = min(max(y0, 0), Hl - 1);
      const int cx1 = min(max(x1, 0), Wl - 1), cy1 = min(max(y1, 0), Hl - 1);
      const float t00 = (okx0 && oky0) ? vb[(size_t)(s0 + cy0 * Wl + cx0) * EMBED] : 0.f;
      const float t10 = (okx1 && oky0) ? vb[(size_t)(s0 + cy0 * Wl + cx1) * EMBED] : 0.f;
      const float t01 = (okx0 && oky1) ? vb[(size_t)(s0 + cy1 * Wl + cx0) * EMBED] : 0.f;
      const float t11 = (okx1 && oky1) ? vb[(size_t)(s0 + cy1 * Wl + cx1) * EMBED] : 0.f;
      const float s = t00 * (1.f - wx) * (1.f - wy) + t10 * wx * (1.f - wy)
                    + t01 * (1.f - wx) * wy         + t11 * wx * wy;
      acc += ap[l * 8 + p] * s;
    }
  }
  attn[(size_t)(cam * NQ + q) * EMBED + h * 32 + lane] = acc;
}

// =====================================================================
// Masked cross-camera aggregation: slots = sum(mask*attn) / max(count,1)
// =====================================================================
__global__ void k_aggregate(const float* __restrict__ attn, const float* __restrict__ maskq,
                            float* __restrict__ slots) {
  const int idx = blockIdx.x * blockDim.x + threadIdx.x;
  if (idx >= NQ * EMBED) return;
  const int q = idx >> 8;
  float s = 0.f, cnt = 0.f;
#pragma unroll
  for (int cam = 0; cam < NUM_CAMS; ++cam) {
    const float mq = maskq[cam * NQ + q];
    if (mq != 0.f) { s += attn[(size_t)cam * NQ * EMBED + idx]; cnt += 1.f; }
  }
  slots[idx] = s / fmaxf(cnt, 1.f);
}

// =====================================================================
// LayerNorm(256) + ReLU. One wave per query, 8 channels/lane, shfl reduce.
// =====================================================================
__global__ void k_lnrelu(const float* __restrict__ x1, const float* __restrict__ g,
                         const float* __restrict__ b, float* __restrict__ x2) {
  const int q = blockIdx.x * (blockDim.x >> 5) + (threadIdx.x >> 5);
  if (q >= NQ) return;
  const int lane = threadIdx.x & 31;
  const float* row = x1 + (size_t)q * EMBED;
  float vals[8];
  float s = 0.f;
#pragma unroll
  for (int e = 0; e < 8; ++e) { vals[e] = row[lane + 32 * e]; s += vals[e]; }
#pragma unroll
  for (int m = 16; m; m >>= 1) s += __shfl_xor(s, m, 32);
  const float mu = s * (1.f / 256.f);
  float vs = 0.f;
#pragma unroll
  for (int e = 0; e < 8; ++e) { const float d = vals[e] - mu; vs += d * d; }
#pragma unroll
  for (int m = 16; m; m >>= 1) vs += __shfl_xor(vs, m, 32);
  const float inv = rsqrtf(vs * (1.f / 256.f) + 1e-5f);
#pragma unroll
  for (int e = 0; e < 8; ++e) {
    const int c = lane + 32 * e;
    const float y = (vals[e] - mu) * inv * g[c] + b[c];
    x2[(size_t)q * EMBED + c] = fmaxf(y, 0.f);
  }
}

// =====================================================================
extern "C" void kernel_launch(void* const* d_in, const int* in_sizes, int n_in,
                              void* d_out, int out_size, void* d_ws, size_t ws_size,
                              hipStream_t stream) {
  (void)in_sizes; (void)n_in; (void)out_size; (void)ws_size;
  const float* feats   = (const float*)d_in[0];   // (6,7729,1,256)
  const float* l2i     = (const float*)d_in[1];   // (1,6,4,4)
  // d_in[2..4]: spatial_shapes / level_start_index / img_shape (compile-time constants)
  const float* bev     = (const float*)d_in[5];   // bev_embed (4096,256)
  const float* pos_w1  = (const float*)d_in[6];
  const float* pos_b1  = (const float*)d_in[7];
  const float* pos_w2  = (const float*)d_in[8];
  const float* pos_b2  = (const float*)d_in[9];
  const float* pos_w3  = (const float*)d_in[10];
  const float* pos_b3  = (const float*)d_in[11];
  const float* off_w   = (const float*)d_in[12];
  const float* off_b   = (const float*)d_in[13];
  const float* attw_w  = (const float*)d_in[14];
  const float* attw_b  = (const float*)d_in[15];
  const float* vproj_w = (const float*)d_in[16];
  const float* vproj_b = (const float*)d_in[17];
  const float* oproj_w = (const float*)d_in[18];
  const float* oproj_b = (const float*)d_in[19];
  const float* ref1_w  = (const float*)d_in[20];
  const float* ref1_b  = (const float*)d_in[21];
  const float* ln_g    = (const float*)d_in[22];
  const float* ln_b    = (const float*)d_in[23];
  const float* ref2_w  = (const float*)d_in[24];
  const float* ref2_b  = (const float*)d_in[25];

  float* ws = (float*)d_ws;
  size_t ofs = 0;
  auto alloc = [&](size_t n) { float* p = ws + ofs; ofs += n; return p; };
  float* pe128 = alloc((size_t)NQ * 128);
  float* pe2   = alloc((size_t)NQ * EMBED);
  float* qp    = alloc((size_t)NQ * EMBED);
  float* offb  = alloc((size_t)NQ * 512);
  float* awb   = alloc((size_t)NQ * 256);
  float* vbuf  = alloc((size_t)NUM_CAMS * NUM_VALUE * EMBED);
  float* refc  = alloc((size_t)NUM_CAMS * NQ * 8);
  float* mkq   = alloc((size_t)NUM_CAMS * NQ);
  float* attn  = alloc((size_t)NUM_CAMS * NQ * EMBED);
  float* slots = alloc((size_t)NQ * EMBED);
  float* s2    = alloc((size_t)NQ * EMBED);
  float* x1    = alloc((size_t)NQ * EMBED);
  float* x2    = alloc((size_t)NQ * EMBED);

  const dim3 blk(256);                              // 8 wave32s / block
  auto gemm_blocks = [](int M, int N) {
    const int tiles = ((M + 31) / 32) * ((N + 31) / 32);
    return (unsigned)((tiles + 7) / 8);
  };

  // positional MLP + query assembly: qp = bev_embed + posMLP(pos2d)
  k_posmlp1<<<(NQ * 128 + 255) / 256, blk, 0, stream>>>(pos_w1, pos_b1, pe128);
  k_gemm<<<gemm_blocks(NQ, 256), blk, 0, stream>>>(pe128, pos_w2, pos_b2, nullptr, pe2, NQ, 256, 128, 1, 0);
  k_gemm<<<gemm_blocks(NQ, 256), blk, 0, stream>>>(pe2, pos_w3, pos_b3, bev, qp, NQ, 256, 256, 0, 0);

  // offsets + attention weights: computed ONCE (identical across cameras)
  k_gemm<<<gemm_blocks(NQ, 512), blk, 0, stream>>>(qp, off_w, off_b, nullptr, offb, NQ, 512, 256, 0, 0);
  k_gemm<<<gemm_blocks(NQ, 256), blk, 0, stream>>>(qp, attw_w, attw_b, nullptr, awb, NQ, 256, 256, 0, 0);
  k_softmax<<<(NQ * NUM_HEADS + 7) / 8, blk, 0, stream>>>(awb);

  // value projection for all cameras as one big GEMM (M = 6*7729)
  k_gemm<<<gemm_blocks(NUM_CAMS * NUM_VALUE, 256), blk, 0, stream>>>(
      feats, vproj_w, vproj_b, nullptr, vbuf, NUM_CAMS * NUM_VALUE, 256, 256, 0, 0);

  // reference points + masks, then the gather-heavy sampling core
  k_refcam<<<(NUM_CAMS * NQ + 255) / 256, blk, 0, stream>>>(l2i, refc, mkq);
  k_sample<<<(NUM_CAMS * NQ * NUM_HEADS + 7) / 8, blk, 0, stream>>>(vbuf, refc, mkq, offb, awb, attn);
  k_aggregate<<<(NQ * EMBED + 255) / 256, blk, 0, stream>>>(attn, mkq, slots);

  // output projection (+bev residual), refine head with LN+ReLU, final N=80 GEMM
  k_gemm<<<gemm_blocks(NQ, 256), blk, 0, stream>>>(slots, oproj_w, oproj_b, bev, s2, NQ, 256, 256, 0, 0);
  k_gemm<<<gemm_blocks(NQ, 256), blk, 0, stream>>>(s2, ref1_w, ref1_b, nullptr, x1, NQ, 256, 256, 0, 0);
  k_lnrelu<<<(NQ + 7) / 8, blk, 0, stream>>>(x1, ln_g, ln_b, x2);
  k_gemm<<<gemm_blocks(NQ, 80), blk, 0, stream>>>(x2, ref2_w, ref2_b, nullptr, (float*)d_out,
                                                  NQ, 80, 256, 0, /*transposed store*/1);
}
